// GaussianFormer_VMR_23141283791537
// MI455X (gfx1250) — matrix operand order repaired and checked by reading, not compiled, not run on hardware
//
#include <hip/hip_runtime.h>
#include <math.h>

typedef __bf16 bf16_t;
typedef __attribute__((ext_vector_type(16))) __bf16 v16bf;
typedef __attribute__((ext_vector_type(8)))  __bf16 v8bf;
typedef __attribute__((ext_vector_type(8)))  float  v8f;

#define B_   32
#define Q_   40
#define L_   1024
#define H_   1024
#define QPAD 48

// ---------------- TDM availability / arity detection ----------------
#if defined(__HIP_DEVICE_COMPILE__) && __has_builtin(__builtin_amdgcn_tensor_load_to_lds)
#define HAVE_TDM 1
#if __has_include(<hip/amd_detail/amd_gfx1250_TDM.h>)
#define TDM_ARITY6 1
#endif
#endif

#if defined(HAVE_TDM)
typedef __attribute__((ext_vector_type(4))) unsigned int u32x4_t;
typedef __attribute__((ext_vector_type(4))) int i32x4_t;
typedef __attribute__((ext_vector_type(8))) int i32x8_t;

// 2D bf16 tile load via Tensor Data Mover. D# layout per CDNA5 ISA 8.3/8.4:
// g0: count=1 | lds_addr | global_addr[56:0] | type=2
// g1: data_size=1(2B), tensor_dim0=tile_w, tensor_dim1=tile_h,
//     tile_dim0=tile_w, tile_dim1=tile_h, tensor_dim0_stride=stride
static __device__ __forceinline__ void tdm_load_tile_bf16(
    void* lds_dst, const void* gsrc,
    unsigned tile_w, unsigned tile_h, unsigned stride_elems) {
  unsigned long long ga = (unsigned long long)(size_t)gsrc;
  unsigned lds = (unsigned)(size_t)lds_dst;   // low 32 bits = LDS byte offset
  u32x4_t g0 = { 1u,                                   // count=1, user mode
                 lds,
                 (unsigned)(ga & 0xffffffffu),
                 (unsigned)((ga >> 32) & 0x01ffffffu) | 0x80000000u };  // type=2
  i32x8_t g1 = { (int)(1u << 16),                                      // data_size=1 (2B)
                 (int)((tile_w & 0xffffu) << 16),                      // tensor_dim0 lo
                 (int)(((tile_w >> 16) & 0xffffu) | ((tile_h & 0xffffu) << 16)),
                 (int)(((tile_h >> 16) & 0xffffu) | ((tile_w & 0xffffu) << 16)),  // | tile_dim0
                 (int)(tile_h & 0xffffu),                              // tile_dim1
                 (int)stride_elems,                                    // dim0_stride lo
                 0, 0 };
  i32x4_t z4 = { 0, 0, 0, 0 };
#if defined(TDM_ARITY6)
  i32x8_t z8 = { 0, 0, 0, 0, 0, 0, 0, 0 };
  __builtin_amdgcn_tensor_load_to_lds(g0, g1, z4, z4, z8, 0);
#else
  __builtin_amdgcn_tensor_load_to_lds(g0, g1, z4, z4, 0);
#endif
}
#endif

static __device__ __forceinline__ unsigned int bfbits(float f) {
  return (unsigned int)__builtin_bit_cast(unsigned short, (bf16_t)f);
}

// ---------------- init: start/end from pred_spans ----------------
__global__ void k_init_spans(const float* __restrict__ pred,
                             float* __restrict__ startP, float* __restrict__ endP) {
  int idx = blockIdx.x * blockDim.x + threadIdx.x;
  if (idx >= B_ * Q_) return;
  float c = pred[idx * 2 + 0], w = pred[idx * 2 + 1];
  startP[idx] = fminf(fmaxf(c - 0.5f * w, 0.0f), 1.0f);
  endP[idx]   = fminf(fmaxf(c + 0.5f * w, 0.0f), 1.0f);
}

// ---------------- txt_off = tanh(txt @ W + b) * 0.5*log_range ----------------
__global__ void k_txt_off(const float* __restrict__ txt, const float* __restrict__ tw,
                          const float* __restrict__ tb, float halflr,
                          float* __restrict__ toff) {
  const int b = blockIdx.x, tid = threadIdx.x;
  float a0 = 0.f, a1 = 0.f;
  for (int h = tid; h < H_; h += 256) {
    float t = txt[(size_t)b * H_ + h];
    a0 += t * tw[h * 2 + 0];
    a1 += t * tw[h * 2 + 1];
  }
  __shared__ float red[512];
  red[tid] = a0; red[256 + tid] = a1;
  __syncthreads();
  for (int o = 128; o > 0; o >>= 1) {
    if (tid < o) { red[tid] += red[tid + o]; red[256 + tid] += red[256 + tid + o]; }
    __syncthreads();
  }
  if (tid == 0) {
    toff[b * 2 + 0] = tanhf(red[0]   + tb[0]) * halflr;
    toff[b * 2 + 1] = tanhf(red[256] + tb[1]) * halflr;
  }
}

// ---------------- fill txt third of joint (bf16) ----------------
__global__ void k_fill_txt(const float* __restrict__ txt, bf16_t* __restrict__ joint) {
  int idx = blockIdx.x * 256 + threadIdx.x;      // < B*Q*H exactly
  int h = idx & (H_ - 1);
  int bq = idx >> 10;
  int b = bq / Q_;
  joint[(size_t)bq * (3 * H_) + 2 * H_ + h] = (bf16_t)txt[(size_t)b * H_ + h];
}

// ---------------- mlp0_w fp32 [3H][H] -> bf16 transposed [H][3H] ----------------
__global__ void k_cvt_w0t(const float* __restrict__ w0, bf16_t* __restrict__ w0t) {
  __shared__ float tile[32][33];
  const int k0 = blockIdx.x * 32;                // 3H/32 = 96
  const int n0 = blockIdx.y * 32;                // H/32  = 32
  const int tx = threadIdx.x & 31, ty = threadIdx.x >> 5;   // 32x8
#pragma unroll
  for (int j = 0; j < 4; ++j) {
    int k = ty + j * 8;
    tile[k][tx] = w0[(size_t)(k0 + k) * H_ + n0 + tx];
  }
  __syncthreads();
#pragma unroll
  for (int j = 0; j < 4; ++j) {
    int n = ty + j * 8;
    w0t[(size_t)(n0 + n) * (3 * H_) + k0 + tx] = (bf16_t)tile[tx][n];
  }
}

// ---------------- Gaussian weights (normalized, bf16, Q padded to 48) ----------------
__global__ void k_weights(const float* __restrict__ startP, const float* __restrict__ endP,
                          const float* __restrict__ toff, const float* __restrict__ vmask,
                          const float* __restrict__ lss, const float* __restrict__ lse,
                          const float* __restrict__ sws, const float* __restrict__ swe,
                          bf16_t* __restrict__ Ws, bf16_t* __restrict__ We) {
  const int qp = blockIdx.x, b = blockIdx.y, tid = threadIdx.x;
  bf16_t* wsrow = Ws + ((size_t)b * QPAD + qp) * L_;
  bf16_t* werow = We + ((size_t)b * QPAD + qp) * L_;
  if (qp >= Q_) {   // uniform per block: zero pad rows so WMMA tiles are clean
#pragma unroll
    for (int i = 0; i < 4; ++i) { int l = tid + i * 256; wsrow[l] = (bf16_t)0.0f; werow[l] = (bf16_t)0.0f; }
    return;
  }
  const int idx = b * Q_ + qp;
  const float s = startP[idx], e = endP[idx];
  const float width = fmaxf(e - s, 1e-6f);
  const float min_sig = 1.0f / (4.0f * (float)L_);
  const float sig_s = fmaxf(__expf(lss[0] + sws[0] * width + toff[b * 2 + 0]), min_sig);
  const float sig_e = fmaxf(__expf(lse[0] + swe[0] * width + toff[b * 2 + 1]), min_sig);

  float vs[4], ve[4];
  float sum_s = 0.f, sum_e = 0.f;
#pragma unroll
  for (int i = 0; i < 4; ++i) {
    const int l = tid + i * 256;
    const float t = (float)l * (1.0f / 1023.0f);
    const float m = vmask[(size_t)b * L_ + l];
    float ds = (t - s) / sig_s; float a = __expf(-0.5f * ds * ds) * m;
    float de = (t - e) / sig_e; float c = __expf(-0.5f * de * de) * m;
    vs[i] = a; ve[i] = c; sum_s += a; sum_e += c;
  }
  __shared__ float red[512];
  red[tid] = sum_s; red[256 + tid] = sum_e;
  __syncthreads();
  for (int o = 128; o > 0; o >>= 1) {
    if (tid < o) { red[tid] += red[tid + o]; red[256 + tid] += red[256 + tid + o]; }
    __syncthreads();
  }
  const float inv_s = 1.0f / fmaxf(red[0],   1e-8f);
  const float inv_e = 1.0f / fmaxf(red[256], 1e-8f);
#pragma unroll
  for (int i = 0; i < 4; ++i) {
    const int l = tid + i * 256;
    wsrow[l] = (bf16_t)(vs[i] * inv_s);
    werow[l] = (bf16_t)(ve[i] * inv_e);
  }
}

// ---------------- pooling GEMM; TDM-staged A panels, transposed B tile ----------------
__global__ void __launch_bounds__(128)
k_pool(const bf16_t* __restrict__ Ws, const bf16_t* __restrict__ We,
       const float* __restrict__ vid, bf16_t* __restrict__ joint) {
  const int h0 = blockIdx.x * 64;
  const int qt = blockIdx.y;
  const int b  = blockIdx.z;
  const int tid  = threadIdx.x;
  const int wave = tid >> 5, lane = tid & 31;
  const int lhalf = lane >> 4, lrow = lane & 15;

  __shared__ bf16_t ApanS[16 * 512];      // 16 KB: A half-panel (start)
  __shared__ bf16_t ApanE[16 * 512];      // 16 KB: A half-panel (end)
  __shared__ unsigned int BtU[1024];      // 4 KB: Bt[64][32] bf16, n-major (fragment-friendly)
  const bf16_t* Bt = reinterpret_cast<const bf16_t*>(BtU);

  const bf16_t* WsBase = Ws + ((size_t)b * QPAD + qt * 16) * L_;
  const bf16_t* WeBase = We + ((size_t)b * QPAD + qt * 16) * L_;
  const float*  VBase  = vid + (size_t)b * L_ * H_ + h0;

  v8f accS = {}; v8f accE = {};
  const int ka = lhalf ? 8 : 0, kh = lhalf ? 24 : 16;   // A K-offsets per ISA 7.12.2
  const int kboff = lhalf * 8;                          // B K-offset
  const int cbn = wave * 16 + lrow;                     // B column (n) for this lane

  for (int half = 0; half < 2; ++half) {
    __syncthreads();                       // previous half fully consumed
#if defined(HAVE_TDM)
    if (tid < 32) {                        // one wave issues both panel DMAs
      tdm_load_tile_bf16(ApanS, WsBase + half * 512, 512u, 16u, (unsigned)L_);
      tdm_load_tile_bf16(ApanE, WeBase + half * 512, 512u, 16u, (unsigned)L_);
      __builtin_amdgcn_s_wait_tensorcnt(0);
    }
#else
#pragma unroll
    for (int it = 0; it < 32; ++it) {      // 4096 uints per panel
      const int i = tid + it * 128;
      const int r = i >> 8, cu = i & 255;  // 16 rows x 256 uints
      ((unsigned int*)ApanS)[i] =
          reinterpret_cast<const unsigned int*>(WsBase + (size_t)r * L_ + half * 512)[cu];
      ((unsigned int*)ApanE)[i] =
          reinterpret_cast<const unsigned int*>(WeBase + (size_t)r * L_ + half * 512)[cu];
    }
#endif
    __syncthreads();

    for (int kb = 0; kb < 16; ++kb) {
      const int l0 = half * 512 + kb * 32;
      const int loff = kb * 32;
      // stage B tile transposed: Bt[n][k], packed pairs (k, k+1) per uint
#pragma unroll
      for (int it = 0; it < 2; ++it) {
        const int i = tid + it * 128;              // 256 tasks: (kpair, nquad)
        const int kp = i >> 4, c4 = i & 15;
        const float4 va = *reinterpret_cast<const float4*>(VBase + (size_t)(l0 + 2 * kp)     * H_ + c4 * 4);
        const float4 vb = *reinterpret_cast<const float4*>(VBase + (size_t)(l0 + 2 * kp + 1) * H_ + c4 * 4);
        BtU[(c4 * 4 + 0) * 16 + kp] = bfbits(va.x) | (bfbits(vb.x) << 16);
        BtU[(c4 * 4 + 1) * 16 + kp] = bfbits(va.y) | (bfbits(vb.y) << 16);
        BtU[(c4 * 4 + 2) * 16 + kp] = bfbits(va.z) | (bfbits(vb.z) << 16);
        BtU[(c4 * 4 + 3) * 16 + kp] = bfbits(va.w) | (bfbits(vb.w) << 16);
      }
      if (l0 + 32 < L_)
        __builtin_prefetch(VBase + (size_t)(l0 + 32) * H_ + tid, 0, 0);
      __syncthreads();

      // vectorized fragment loads: 2x b128 per matrix per lane
      v8bf aS0 = *reinterpret_cast<const v8bf*>(ApanS + lrow * 512 + loff + ka);
      v8bf aS1 = *reinterpret_cast<const v8bf*>(ApanS + lrow * 512 + loff + kh);
      v8bf aE0 = *reinterpret_cast<const v8bf*>(ApanE + lrow * 512 + loff + ka);
      v8bf aE1 = *reinterpret_cast<const v8bf*>(ApanE + lrow * 512 + loff + kh);
      v8bf bl  = *reinterpret_cast<const v8bf*>(Bt + cbn * 32 + kboff);
      v8bf bh  = *reinterpret_cast<const v8bf*>(Bt + cbn * 32 + kboff + 16);
      v16bf aS = __builtin_shufflevector(aS0, aS1, 0,1,2,3,4,5,6,7,8,9,10,11,12,13,14,15);
      v16bf aE = __builtin_shufflevector(aE0, aE1, 0,1,2,3,4,5,6,7,8,9,10,11,12,13,14,15);
      v16bf bF = __builtin_shufflevector(bl,  bh,  0,1,2,3,4,5,6,7,8,9,10,11,12,13,14,15);

      accS = __builtin_amdgcn_wmma_f32_16x16x32_bf16(false, aS, false, bF, (short)0, accS, false, false);
      accE = __builtin_amdgcn_wmma_f32_16x16x32_bf16(false, aE, false, bF, (short)0, accE, false, false);
      __syncthreads();
    }
  }

#pragma unroll
  for (int r = 0; r < 8; ++r) {
    const int q = qt * 16 + r + 8 * lhalf;
    if (q < Q_) {
      const size_t rowoff = (size_t)(b * Q_ + q) * (3 * H_);
      const int col = h0 + wave * 16 + lrow;
      joint[rowoff + col]      = (bf16_t)accS[r];
      joint[rowoff + H_ + col] = (bf16_t)accE[r];
    }
  }
}

// ---------------- MLP0 GEMM: hidden = relu(joint @ W0 + b0), B from w0t ----------------
__global__ void __launch_bounds__(128)
k_mlp0(const bf16_t* __restrict__ joint, const bf16_t* __restrict__ w0t,
       const float* __restrict__ b0, float* __restrict__ hidden) {
  const int n0 = blockIdx.x * 64;
  const int mt = blockIdx.y;          // 0..79 over 1280 flattened rows
  const int tid  = threadIdx.x;
  const int wave = tid >> 5, lane = tid & 31;
  const int lhalf = lane >> 4, lrow = lane & 15;

  __shared__ unsigned int AU[256];    // A tile 16x32 bf16, row-major
  __shared__ unsigned int BtU[1024];  // Bt[64][32] bf16, n-major
  const bf16_t* As = reinterpret_cast<const bf16_t*>(AU);
  const bf16_t* Bt = reinterpret_cast<const bf16_t*>(BtU);

  const bf16_t* ABase = joint + (size_t)mt * 16 * (3 * H_);
  const bf16_t* BBase = w0t + (size_t)n0 * (3 * H_);
  v8f acc = {};
  const int ka = lhalf ? 8 : 0, kh = lhalf ? 24 : 16;
  const int kboff = lhalf * 8;
  const int cbn = wave * 16 + lrow;

  for (int kb = 0; kb < (3 * H_) / 32; ++kb) {
    const int k0 = kb * 32;
#pragma unroll
    for (int it = 0; it < 2; ++it) {
      const int i = tid + it * 128;
      const int r = i >> 4, cu = i & 15;
      AU[i] = reinterpret_cast<const unsigned int*>(ABase + (size_t)r * (3 * H_) + k0)[cu];
    }
#pragma unroll
    for (int it = 0; it < 8; ++it) {
      const int i = tid + it * 128;
      const int r = i >> 4, cu = i & 15;   // 64 n-rows x 16 uints, contiguous copy
      BtU[i] = reinterpret_cast<const unsigned int*>(BBase + (size_t)r * (3 * H_) + k0)[cu];
    }
    __syncthreads();

    v8bf a0 = *reinterpret_cast<const v8bf*>(As + lrow * 32 + ka);
    v8bf a1 = *reinterpret_cast<const v8bf*>(As + lrow * 32 + kh);
    v8bf bl = *reinterpret_cast<const v8bf*>(Bt + cbn * 32 + kboff);
    v8bf bh = *reinterpret_cast<const v8bf*>(Bt + cbn * 32 + kboff + 16);
    v16bf a  = __builtin_shufflevector(a0, a1, 0,1,2,3,4,5,6,7,8,9,10,11,12,13,14,15);
    v16bf bF = __builtin_shufflevector(bl, bh, 0,1,2,3,4,5,6,7,8,9,10,11,12,13,14,15);
    acc = __builtin_amdgcn_wmma_f32_16x16x32_bf16(false, a, false, bF, (short)0, acc, false, false);
    __syncthreads();
  }
  const int col = n0 + wave * 16 + lrow;
  const float bias = b0[col];
#pragma unroll
  for (int r = 0; r < 8; ++r) {
    const int row = mt * 16 + r + 8 * lhalf;
    hidden[(size_t)row * H_ + col] = fmaxf(acc[r] + bias, 0.0f);
  }
}

// ---------------- MLP1 head + span update + outputs ----------------
__global__ void k_mlp1(const float* __restrict__ hidden, const float* __restrict__ w1,
                       const float* __restrict__ b1, float* __restrict__ startP,
                       float* __restrict__ endP, float* __restrict__ out,
                       int pass, int last) {
  const int bq = blockIdx.x, tid = threadIdx.x;
  const float* hrow = hidden + (size_t)bq * H_;
  float a0 = 0.f, a1 = 0.f;
  for (int h = tid; h < H_; h += 256) {
    float v = hrow[h];
    a0 += v * w1[h * 2 + 0];
    a1 += v * w1[h * 2 + 1];
  }
  __shared__ float red[512];
  red[tid] = a0; red[256 + tid] = a1;
  __syncthreads();
  for (int o = 128; o > 0; o >>= 1) {
    if (tid < o) { red[tid] += red[tid + o]; red[256 + tid] += red[256 + tid + o]; }
    __syncthreads();
  }
  if (tid == 0) {
    float d0 = tanhf(red[0]   + b1[0]) * 0.1f;
    float d1 = tanhf(red[256] + b1[1]) * 0.1f;
    float s = fminf(fmaxf(startP[bq] + d0, 0.0f), 1.0f);
    float e = fminf(fmaxf(endP[bq]   + d1, 0.0f), 1.0f);
    startP[bq] = s; endP[bq] = e;
    float c = 0.5f * (s + e);
    float w = fmaxf(e - s, 1e-6f);
    const int stride = B_ * Q_ * 2;
    out[stride + pass * stride + bq * 2 + 0] = c;   // stacked passes
    out[stride + pass * stride + bq * 2 + 1] = w;
    if (last) { out[bq * 2 + 0] = c; out[bq * 2 + 1] = w; }   // final result
  }
}

extern "C" void kernel_launch(void* const* d_in, const int* in_sizes, int n_in,
                              void* d_out, int out_size, void* d_ws, size_t ws_size,
                              hipStream_t stream) {
  (void)in_sizes; (void)n_in; (void)out_size; (void)ws_size;
  const float* pred  = (const float*)d_in[0];
  const float* vid   = (const float*)d_in[1];
  const float* vmask = (const float*)d_in[2];
  const float* txt   = (const float*)d_in[3];
  const float* lss   = (const float*)d_in[4];
  const float* lse   = (const float*)d_in[5];
  const float* sws   = (const float*)d_in[6];
  const float* swe   = (const float*)d_in[7];
  const float* tw    = (const float*)d_in[8];
  const float* tb    = (const float*)d_in[9];
  const float* w0    = (const float*)d_in[10];
  const float* b0    = (const float*)d_in[11];
  const float* w1    = (const float*)d_in[12];
  const float* b1    = (const float*)d_in[13];
  float* out = (float*)d_out;

  char* base = (char*)d_ws;
  size_t off = 0;
  auto take = [&](size_t bytes) -> void* {
    void* p = base + off;
    off = (off + bytes + 255) & ~(size_t)255;
    return p;
  };
  float*  startP = (float*) take((size_t)B_ * Q_ * sizeof(float));
  float*  endP   = (float*) take((size_t)B_ * Q_ * sizeof(float));
  float*  toff   = (float*) take((size_t)B_ * 2  * sizeof(float));
  bf16_t* Ws     = (bf16_t*)take((size_t)B_ * QPAD * L_ * 2);
  bf16_t* We     = (bf16_t*)take((size_t)B_ * QPAD * L_ * 2);
  bf16_t* joint  = (bf16_t*)take((size_t)B_ * Q_ * 3 * H_ * 2);
  bf16_t* w0t    = (bf16_t*)take((size_t)3 * H_ * H_ * 2);
  float*  hidden = (float*) take((size_t)B_ * Q_ * H_ * sizeof(float));

  const float halflr = 0.5f * logf(1.0f + 1e-6f);   // 0.5 * log_range

  k_init_spans<<<dim3((B_ * Q_ + 255) / 256), dim3(256), 0, stream>>>(pred, startP, endP);
  k_txt_off<<<dim3(B_), dim3(256), 0, stream>>>(txt, tw, tb, halflr, toff);
  k_fill_txt<<<dim3((B_ * Q_ * H_) / 256), dim3(256), 0, stream>>>(txt, joint);
  k_cvt_w0t<<<dim3((3 * H_) / 32, H_ / 32), dim3(256), 0, stream>>>(w0, w0t);

  for (int pass = 0; pass < 2; ++pass) {
    k_weights<<<dim3(QPAD, B_), dim3(256), 0, stream>>>(startP, endP, toff, vmask,
                                                        lss, lse, sws, swe, Ws, We);
    k_pool<<<dim3(H_ / 64, 3, B_), dim3(128), 0, stream>>>(Ws, We, vid, joint);
    k_mlp0<<<dim3(H_ / 64, (B_ * Q_) / 16), dim3(128), 0, stream>>>(joint, w0t, b0, hidden);
    k_mlp1<<<dim3(B_ * Q_), dim3(256), 0, stream>>>(hidden, w1, b1, startP, endP, out,
                                                    pass, pass == 1 ? 1 : 0);
  }
}
